// AttentionOpt_57226144252116
// MI455X (gfx1250) — matrix operand order repaired and checked by reading, not compiled, not run on hardware
//
#include <hip/hip_runtime.h>

// ---------------- MI455X / gfx1250 attention (bf16 WMMA, f32 softmax) --------
// B=4 N=2048 H=4 D=32 ALL=128. Memory-bound on bias streaming (128MB, L2-resident);
// all matmuls use v_wmma_f32_16x16x32_bf16. Weights pre-transposed/converted to
// bf16 once so GEMM inner loops are pure b128 loads + WMMA (no per-iter cvt VALU).

constexpr int Bc = 4, Nc = 2048, Hc = 4, Dc = 32, ALLD = 128, OUTc = 128;

typedef __attribute__((ext_vector_type(16))) __bf16    v16bf;
typedef __attribute__((ext_vector_type(8)))  float     v8f;
typedef __attribute__((ext_vector_type(4)))  unsigned  u32x4;
typedef __attribute__((ext_vector_type(4)))  float     f32x4;

union ABFrag { v16bf v; u32x4 q[2]; unsigned short s[16]; };
union CFrag  { v8f   v; float f[8]; };

__device__ inline unsigned short f2bf(float x) {
  union { float f; unsigned u; } c; c.f = x;
  unsigned r = c.u + 0x7FFFu + ((c.u >> 16) & 1u);   // round-to-nearest-even
  return (unsigned short)(r >> 16);
}

__device__ inline v8f wmma_bf16(const ABFrag& a, const ABFrag& b, v8f c) {
  // D = A(16x32 bf16) * B(32x16 bf16) + C(16x16 f32)
  return __builtin_amdgcn_wmma_f32_16x16x32_bf16(false, a.v, false, b.v,
                                                 (short)0, c, false, false);
}

// ---------------------------------------------------------------------------
// Kernel 0: one-time weight transpose + bf16 convert.
// Wt[m][n*128 + k] = bf16(W_m[k][n]),  m: 0=Wq 1=Wk 2=Wv 3=Wg 4=Wo.
// B-fragment for col n then = 16 contiguous bf16 at Wt[m] + n*128 + 16*lh.
// ---------------------------------------------------------------------------
__global__ __launch_bounds__(256) void wconv_kernel(
    const float* __restrict__ Wq, const float* __restrict__ Wk,
    const float* __restrict__ Wv, const float* __restrict__ Wg,
    const float* __restrict__ Wo, unsigned short* __restrict__ Wt)
{
  const int m = blockIdx.x;
  const float* W = (m == 0) ? Wq : (m == 1) ? Wk : (m == 2) ? Wv : (m == 3) ? Wg : Wo;
  unsigned short* T = Wt + (size_t)m * ALLD * ALLD;
  for (int i = threadIdx.x; i < ALLD * ALLD; i += 256) {
    const int n = i >> 7, k = i & 127;
    T[i] = f2bf(W[(size_t)k * ALLD + n]);
  }
}

// ---------------------------------------------------------------------------
// Kernel 1: projections.  blockIdx.y: 0=Q 1=K 2=V 3=gate.
// Each block: 16 rows x 128 cols; 8 waves, wave w -> cols [16w,16w+16).
// ---------------------------------------------------------------------------
__global__ __launch_bounds__(256) void proj_kernel(
    const float* __restrict__ qd, const float* __restrict__ md,
    const unsigned short* __restrict__ Wt, const float* __restrict__ bg,
    unsigned short* __restrict__ Qb, unsigned short* __restrict__ Kb,
    unsigned short* __restrict__ Vt, float* __restrict__ gate)
{
  const int which = blockIdx.y;
  const float* X = (which == 1 || which == 2) ? md : qd;
  const unsigned short* W = Wt + (size_t)which * ALLD * ALLD;   // bf16, transposed
  const int r0 = blockIdx.x * 16;
  const int bi = r0 / Nc, nq0 = r0 % Nc;
  const int tid = threadIdx.x, wave = tid >> 5, lane = tid & 31;
  const int lh = lane >> 4, lm = lane & 15;

  __shared__ unsigned short sA[16][ALLD + 8];        // bf16 input tile
  for (int i = tid; i < 16 * ALLD / 4; i += 256) {
    const int row = i >> 5, col = (i & 31) << 2;
    const f32x4 xv = *reinterpret_cast<const f32x4*>(X + (size_t)(r0 + row) * ALLD + col);
    sA[row][col + 0] = f2bf(xv.x); sA[row][col + 1] = f2bf(xv.y);
    sA[row][col + 2] = f2bf(xv.z); sA[row][col + 3] = f2bf(xv.w);
  }
  __syncthreads();

  const int n0 = wave * 16;
  CFrag c;
  #pragma unroll
  for (int j = 0; j < 8; ++j) c.f[j] = 0.f;

  #pragma unroll
  for (int kk = 0; kk < ALLD; kk += 32) {
    ABFrag a, b;
    // A-layout: lane M=lm, K = kk+16*lh+[0..15]
    const unsigned short* ar = &sA[lm][kk + 16 * lh];
    a.q[0] = *reinterpret_cast<const u32x4*>(ar);
    a.q[1] = *reinterpret_cast<const u32x4*>(ar + 8);
    // B-layout: lane N=n0+lm, K = kk+16*lh+[0..15] -> contiguous in Wt
    const unsigned short* wr = W + (size_t)(n0 + lm) * ALLD + kk + 16 * lh;
    b.q[0] = *reinterpret_cast<const u32x4*>(wr);
    b.q[1] = *reinterpret_cast<const u32x4*>(wr + 8);
    c.v = wmma_bf16(a, b, c.v);
  }

  const int col = n0 + lm;
  #pragma unroll
  for (int j = 0; j < 8; ++j) {                       // C-layout: M = j+8*lh, N = lm
    const int row = nq0 + j + 8 * lh;
    const float val = c.f[j];
    if (which == 3) {
      const float g = 1.0f / (1.0f + __expf(-(val + bg[col])));
      gate[((size_t)bi * Nc + row) * ALLD + col] = g;
    } else {
      const int h = col >> 5, d = col & 31;
      const unsigned short bv = f2bf(val);
      const size_t qk = (((size_t)bi * Hc + h) * Nc + row) * Dc + d;
      if (which == 0)      Qb[qk] = bv;
      else if (which == 1) Kb[qk] = bv;
      else                 Vt[(((size_t)bi * Hc + h) * Dc + d) * Nc + row] = bv;
    }
  }
}

// ---------------------------------------------------------------------------
// Kernel 2: flash attention. Block = 32 queries of one (b,h). 8 waves split the
// key dim (256 keys each, online softmax), then LSE-merge via LDS ds_add_f32.
// ---------------------------------------------------------------------------
__global__ __launch_bounds__(256) void attn_kernel(
    const unsigned short* __restrict__ Qb, const unsigned short* __restrict__ Kb,
    const unsigned short* __restrict__ Vt,
    const float* __restrict__ bias, const float* __restrict__ nbias,
    float* __restrict__ wa)
{
  const int q0 = blockIdx.x * 32;
  const int bh = blockIdx.y, bi = bh / Hc, h = bh % Hc;
  const int tid = threadIdx.x, wave = tid >> 5, lane = tid & 31;
  const int lh = lane >> 4, lm = lane & 15;

  const unsigned short* Qp = Qb + (size_t)bh * Nc * Dc;
  const unsigned short* Kp = Kb + (size_t)bh * Nc * Dc;
  const unsigned short* Vp = Vt + (size_t)bh * Dc * Nc;
  const float* bp = bias  + ((size_t)bi * Nc + q0) * Nc;
  const float* np = nbias + ((size_t)h  * Nc + q0) * Nc;

  __shared__ __align__(16) unsigned char smem[36 * 1024];
  unsigned short* sP  = reinterpret_cast<unsigned short*>(smem);       // 16KB: [8][2][16][32]
  float*          smf = reinterpret_cast<float*>(smem);                // 16KB (after loop)
  float*          slf = reinterpret_cast<float*>(smem + 16 * 1024);    // 16KB
  float*          sof = reinterpret_cast<float*>(smem + 32 * 1024);    //  4KB

  // Q A-fragments for the two 16-query subtiles
  ABFrag a[2];
  #pragma unroll
  for (int qt = 0; qt < 2; ++qt) {
    const unsigned short* qr = Qp + (size_t)(q0 + 16 * qt + lm) * Dc + 16 * lh;
    a[qt].q[0] = *reinterpret_cast<const u32x4*>(qr);
    a[qt].q[1] = *reinterpret_cast<const u32x4*>(qr + 8);
  }

  float mrow[2][8], lrow[2][8];
  CFrag acc[2][2];
  #pragma unroll
  for (int qt = 0; qt < 2; ++qt)
    #pragma unroll
    for (int j = 0; j < 8; ++j) {
      mrow[qt][j] = -1e30f; lrow[qt][j] = 0.f;
      acc[qt][0].f[j] = 0.f; acc[qt][1].f[j] = 0.f;
    }

  const int kBeg = wave * (Nc / 8), kEnd = kBeg + (Nc / 8);
  for (int kc = kBeg; kc < kEnd; kc += 32) {
    if (kc + 32 < kEnd) {                              // global_prefetch_b8 next bias rows
      __builtin_prefetch(bp + (size_t)lane * Nc + kc + 32, 0, 0);
      __builtin_prefetch(np + (size_t)lane * Nc + kc + 32, 0, 0);
    }
    // K^T B-fragments: lane col n=key kc+16t+lm, K-dim d = 16*lh+[0..15]
    ABFrag kb[2];
    #pragma unroll
    for (int t = 0; t < 2; ++t) {
      const unsigned short* kr = Kp + (size_t)(kc + 16 * t + lm) * Dc + 16 * lh;
      kb[t].q[0] = *reinterpret_cast<const u32x4*>(kr);
      kb[t].q[1] = *reinterpret_cast<const u32x4*>(kr + 8);
    }
    // scores
    CFrag s[2][2];
    #pragma unroll
    for (int qt = 0; qt < 2; ++qt)
      #pragma unroll
      for (int t = 0; t < 2; ++t) {
        CFrag z;
        #pragma unroll
        for (int j = 0; j < 8; ++j) z.f[j] = 0.f;
        s[qt][t].v = wmma_bf16(a[qt], kb[t], z.v);
      }
    // + bias + nonbatched_bias (f32, the bandwidth-dominant stream)
    #pragma unroll
    for (int qt = 0; qt < 2; ++qt)
      #pragma unroll
      for (int t = 0; t < 2; ++t)
        #pragma unroll
        for (int j = 0; j < 8; ++j) {
          const size_t ro = (size_t)(16 * qt + j + 8 * lh) * Nc + (kc + 16 * t + lm);
          s[qt][t].f[j] += bp[ro] + np[ro];
        }
    // online softmax over this 32-key chunk
    #pragma unroll
    for (int qt = 0; qt < 2; ++qt) {
      float al[8];
      #pragma unroll
      for (int j = 0; j < 8; ++j) {
        float v = fmaxf(s[qt][0].f[j], s[qt][1].f[j]);
        #pragma unroll
        for (int o = 1; o < 16; o <<= 1) v = fmaxf(v, __shfl_xor(v, o, 32));
        const float mn = fmaxf(mrow[qt][j], v);
        al[j] = __expf(mrow[qt][j] - mn);
        mrow[qt][j] = mn;
      }
      #pragma unroll
      for (int j = 0; j < 8; ++j) {
        s[qt][0].f[j] = __expf(s[qt][0].f[j] - mrow[qt][j]);
        s[qt][1].f[j] = __expf(s[qt][1].f[j] - mrow[qt][j]);
        float rs = s[qt][0].f[j] + s[qt][1].f[j];
        #pragma unroll
        for (int o = 1; o < 16; o <<= 1) rs += __shfl_xor(rs, o, 32);
        lrow[qt][j] = lrow[qt][j] * al[j] + rs;
        acc[qt][0].f[j] *= al[j];
        acc[qt][1].f[j] *= al[j];
      }
      // stage P (C-layout -> row-major [16][32] bf16) in per-wave LDS buffer
      unsigned short* pbuf = sP + (size_t)(wave * 2 + qt) * 16 * 32;
      #pragma unroll
      for (int t = 0; t < 2; ++t)
        #pragma unroll
        for (int j = 0; j < 8; ++j)
          pbuf[(j + 8 * lh) * 32 + 16 * t + lm] = f2bf(s[qt][t].f[j]);
    }
    // V B-fragments from transposed V: lane col d = 16*dsub+lm, K-dim key contiguous
    ABFrag vbf[2];
    #pragma unroll
    for (int d2 = 0; d2 < 2; ++d2) {
      const unsigned short* vr = Vp + (size_t)(16 * d2 + lm) * Nc + kc + 16 * lh;
      vbf[d2].q[0] = *reinterpret_cast<const u32x4*>(vr);
      vbf[d2].q[1] = *reinterpret_cast<const u32x4*>(vr + 8);
    }
    // P·V  (DS ops in-order within wave -> staging safe without barrier)
    #pragma unroll
    for (int qt = 0; qt < 2; ++qt) {
      ABFrag pA;
      const unsigned short* pr = sP + ((size_t)(wave * 2 + qt) * 16 + lm) * 32 + 16 * lh;
      pA.q[0] = *reinterpret_cast<const u32x4*>(pr);
      pA.q[1] = *reinterpret_cast<const u32x4*>(pr + 8);
      acc[qt][0].v = wmma_bf16(pA, vbf[0], acc[qt][0].v);
      acc[qt][1].v = wmma_bf16(pA, vbf[1], acc[qt][1].v);
    }
  }

  // ---- cross-wave log-sum-exp merge ----
  __syncthreads();
  #pragma unroll
  for (int qt = 0; qt < 2; ++qt)
    #pragma unroll
    for (int j = 0; j < 8; ++j) {
      smf[(((size_t)wave * 2 + qt) * 32 + lane) * 8 + j] = mrow[qt][j];
      slf[(((size_t)wave * 2 + qt) * 32 + lane) * 8 + j] = lrow[qt][j];
    }
  for (int i = tid; i < 1024; i += 256) sof[i] = 0.f;
  __syncthreads();
  #pragma unroll
  for (int qt = 0; qt < 2; ++qt)
    #pragma unroll
    for (int j = 0; j < 8; ++j) {
      float M = -1e30f;
      #pragma unroll
      for (int w = 0; w < 8; ++w)
        M = fmaxf(M, smf[(((size_t)w * 2 + qt) * 32 + lane) * 8 + j]);
      const float sc = __expf(mrow[qt][j] - M);
      atomicAdd(&sof[((qt * 2 + 0) * 32 + lane) * 8 + j], acc[qt][0].f[j] * sc);
      atomicAdd(&sof[((qt * 2 + 1) * 32 + lane) * 8 + j], acc[qt][1].f[j] * sc);
    }
  __syncthreads();
  if (wave < 2) {
    const int qt = wave;
    #pragma unroll
    for (int j = 0; j < 8; ++j) {
      float M = -1e30f;
      #pragma unroll
      for (int w = 0; w < 8; ++w)
        M = fmaxf(M, smf[(((size_t)w * 2 + qt) * 32 + lane) * 8 + j]);
      float L = 0.f;
      #pragma unroll
      for (int w = 0; w < 8; ++w) {
        const float mw = smf[(((size_t)w * 2 + qt) * 32 + lane) * 8 + j];
        L += slf[(((size_t)w * 2 + qt) * 32 + lane) * 8 + j] * __expf(mw - M);
      }
      const float inv = 1.0f / L;
      const int row = q0 + 16 * qt + j + 8 * lh;
      #pragma unroll
      for (int d2 = 0; d2 < 2; ++d2) {
        const float o = sof[((qt * 2 + d2) * 32 + lane) * 8 + j] * inv;
        wa[((size_t)bi * Nc + row) * ALLD + h * Dc + d2 * 16 + lm] = o;
      }
    }
  }
}

// ---------------------------------------------------------------------------
// Kernel 3: out = (wa * gate) @ Wo + bo
// ---------------------------------------------------------------------------
__global__ __launch_bounds__(256) void outproj_kernel(
    const float* __restrict__ wa, const float* __restrict__ gate,
    const unsigned short* __restrict__ Wt, const float* __restrict__ bo,
    float* __restrict__ out)
{
  const unsigned short* W = Wt + (size_t)4 * ALLD * ALLD;   // transposed bf16 Wo
  const int r0 = blockIdx.x * 16;
  const int tid = threadIdx.x, wave = tid >> 5, lane = tid & 31;
  const int lh = lane >> 4, lm = lane & 15;

  __shared__ unsigned short sA[16][ALLD + 8];
  for (int i = tid; i < 16 * ALLD / 4; i += 256) {
    const int row = i >> 5, col = (i & 31) << 2;
    const size_t base = (size_t)(r0 + row) * ALLD + col;
    const f32x4 xv = *reinterpret_cast<const f32x4*>(wa + base);
    const f32x4 gv = *reinterpret_cast<const f32x4*>(gate + base);
    sA[row][col + 0] = f2bf(xv.x * gv.x); sA[row][col + 1] = f2bf(xv.y * gv.y);
    sA[row][col + 2] = f2bf(xv.z * gv.z); sA[row][col + 3] = f2bf(xv.w * gv.w);
  }
  __syncthreads();

  const int n0 = wave * 16;
  CFrag c;
  #pragma unroll
  for (int j = 0; j < 8; ++j) c.f[j] = 0.f;

  #pragma unroll
  for (int kk = 0; kk < ALLD; kk += 32) {
    ABFrag a, b;
    const unsigned short* ar = &sA[lm][kk + 16 * lh];
    a.q[0] = *reinterpret_cast<const u32x4*>(ar);
    a.q[1] = *reinterpret_cast<const u32x4*>(ar + 8);
    const unsigned short* wr = W + (size_t)(n0 + lm) * OUTc + kk + 16 * lh;
    b.q[0] = *reinterpret_cast<const u32x4*>(wr);
    b.q[1] = *reinterpret_cast<const u32x4*>(wr + 8);
    c.v = wmma_bf16(a, b, c.v);
  }
  const float bv = bo[n0 + lm];
  #pragma unroll
  for (int j = 0; j < 8; ++j)
    out[(size_t)(r0 + j + 8 * lh) * OUTc + n0 + lm] = c.f[j] + bv;
}

// ---------------------------------------------------------------------------
extern "C" void kernel_launch(void* const* d_in, const int* in_sizes, int n_in,
                              void* d_out, int out_size, void* d_ws, size_t ws_size,
                              hipStream_t stream) {
  (void)in_sizes; (void)n_in; (void)out_size; (void)ws_size;
  const float* qd   = (const float*)d_in[0];
  const float* md   = (const float*)d_in[1];
  const float* bias = (const float*)d_in[2];
  const float* nb   = (const float*)d_in[3];
  const float* Wq   = (const float*)d_in[4];
  const float* Wk   = (const float*)d_in[5];
  const float* Wv   = (const float*)d_in[6];
  const float* Wg   = (const float*)d_in[7];
  const float* bg   = (const float*)d_in[8];
  const float* Wo   = (const float*)d_in[9];
  const float* bo   = (const float*)d_in[10];
  float* out = (float*)d_out;

  unsigned char* ws = (unsigned char*)d_ws;
  const size_t MB = 1024 * 1024;
  unsigned short* Qb = (unsigned short*)(ws + 0);        // bf16 [B,H,N,D]  2MB
  unsigned short* Kb = (unsigned short*)(ws + 2 * MB);   // bf16 [B,H,N,D]  2MB
  unsigned short* Vt = (unsigned short*)(ws + 4 * MB);   // bf16 [B,H,D,N]  2MB
  float* gate        = (float*)(ws + 6 * MB);            // f32  [B,N,128]  4MB
  float* wa          = (float*)(ws + 10 * MB);           // f32  [B,N,128]  4MB
  unsigned short* Wt = (unsigned short*)(ws + 14 * MB);  // bf16 [5][128][128] 160KB

  wconv_kernel<<<5, 256, 0, stream>>>(Wq, Wk, Wv, Wg, Wo, Wt);
  proj_kernel<<<dim3((Bc * Nc) / 16, 4), 256, 0, stream>>>(qd, md, Wt, bg,
                                                           Qb, Kb, Vt, gate);
  attn_kernel<<<dim3(Nc / 32, Bc * Hc), 256, 0, stream>>>(Qb, Kb, Vt, bias, nb, wa);
  outproj_kernel<<<(Bc * Nc) / 16, 256, 0, stream>>>(wa, gate, Wt, bo, out);
}
